// Attention_12146167513140
// MI455X (gfx1250) — matrix operand order repaired and checked by reading, not compiled, not run on hardware
//
#include <hip/hip_runtime.h>
#include <hip/hip_bf16.h>

typedef __attribute__((ext_vector_type(16))) _Float16 v16h;
typedef __attribute__((ext_vector_type(8)))  _Float16 v8h;
typedef __attribute__((ext_vector_type(8)))  float    v8f;

#define BB   2
#define N1N  3072
#define N2N  3072
#define CS   256
#define HN   8
#define DN   32
#define HC   256
#define INFV 100000.0f
#define EPSF 1e-8f
// sqrt(1/(3*D)) = sqrt(1/96), folded into stored Q
#define SCALEQ 0.10206207261596575f

static __device__ __forceinline__ v16h pack16(const _Float16* lo_p, const _Float16* hi_p) {
  const v8h lo = *(const v8h*)lo_p;
  const v8h hi = *(const v8h*)hi_p;
  v16h r;
#pragma unroll
  for (int i = 0; i < 8; ++i) { r[i] = lo[i]; r[8 + i] = hi[i]; }
  return r;
}

// ---------------------------------------------------------------------------
// s1e = f16( s1 + sinusoidal_embed(ridx1 - ridx1[ct1], 256) )
// grid: B*N1 blocks x 256 threads (one row per block, one channel per thread)
__global__ void embed_kernel(const float* __restrict__ s1,
                             const int* __restrict__ ridx1,
                             const int* __restrict__ ct1,
                             _Float16* __restrict__ s1e) {
  const int bi = blockIdx.x;            // b*N1 + i
  const int c  = threadIdx.x;           // 0..255
  const int b  = bi / N1N;
  const int ct = ct1[b];
  const int pos = ridx1[bi] - ridx1[b * N1N + ct];
  const int j = c & 127;
  const float fr  = __expf(-9.210340371976184f * (float)j * (1.0f / 128.0f));
  const float ang = (float)pos * fr;
  const float e   = (c < 128) ? __sinf(ang) : __cosf(ang);
  s1e[(size_t)bi * CS + c] = (_Float16)(s1[(size_t)bi * CS + c] + e);
}

// dst[i] = f16(src[i])
__global__ void cvt_kernel(const float* __restrict__ src, _Float16* __restrict__ dst, int n) {
  const int i = blockIdx.x * blockDim.x + threadIdx.x;
  if (i < n) dst[i] = (_Float16)src[i];
}

// dst[n*K + k] = f16(src[k*N + n])   (weight transpose+convert, N-major for B fragments)
__global__ void tcvt_kernel(const float* __restrict__ src, _Float16* __restrict__ dst, int K, int N) {
  const int i = blockIdx.x * blockDim.x + threadIdx.x;
  if (i < K * N) {
    const int k = i / N, n = i - k * N;
    dst[(size_t)n * K + k] = (_Float16)src[i];
  }
}

// ---------------------------------------------------------------------------
// Q projection + RMS-norm + scale fold.  grid: B*N1/16, block 256 (8 waves = 8 heads)
__global__ __launch_bounds__(256) void qproj_kernel(const _Float16* __restrict__ s1e,
                                                    const _Float16* __restrict__ WqT,
                                                    const float* __restrict__ gq,
                                                    _Float16* __restrict__ qf) {
  const int tile = blockIdx.x;
  const int b    = tile / (N1N / 16);
  const int row0 = (tile - b * (N1N / 16)) * 16;
  const int lane = threadIdx.x & 31;
  const int head = threadIdx.x >> 5;
  const int hgrp = lane >> 4;
  const int lid  = lane & 15;

  const _Float16* arow = s1e + ((size_t)b * N1N + row0 + lid) * CS;
  v8f c0 = {}, c1 = {};
  for (int kk = 0; kk < CS; kk += 32) {
    const v16h a = pack16(arow + kk + hgrp * 16, arow + kk + hgrp * 16 + 8);
    {
      const _Float16* bp = WqT + (size_t)((head * 2 + 0) * 16 + lid) * CS + kk + hgrp * 16;
      const v16h bf = pack16(bp, bp + 8);
      c0 = __builtin_amdgcn_wmma_f32_16x16x32_f16(false, a, false, bf, (short)0, c0, false, false);
    }
    {
      const _Float16* bp = WqT + (size_t)((head * 2 + 1) * 16 + lid) * CS + kk + hgrp * 16;
      const v16h bf = pack16(bp, bp + 8);
      c1 = __builtin_amdgcn_wmma_f32_16x16x32_f16(false, a, false, bf, (short)0, c1, false, false);
    }
  }
  const float gql = gq[lid], gqh = gq[16 + lid];
#pragma unroll
  for (int r = 0; r < 8; ++r) {
    float p = c0[r] * c0[r] + c1[r] * c1[r];
    p += __shfl_xor(p, 1); p += __shfl_xor(p, 2); p += __shfl_xor(p, 4); p += __shfl_xor(p, 8);
    const float rinv = rsqrtf(p * (1.0f / 32.0f) + EPSF) * SCALEQ;
    const int m = hgrp * 8 + r;
    const size_t base = (((size_t)b * HN + head) * N1N + (row0 + m)) * (size_t)DN;
    qf[base + lid]      = (_Float16)(c0[r] * rinv * gql);
    qf[base + 16 + lid] = (_Float16)(c1[r] * rinv * gqh);
  }
}

// KV projection; K gets RMS-norm, V is stored transposed (B,H,D,N2).
// grid: B*N2/16, block 256 (wave = head, 4 n-tiles each: 2 K + 2 V)
__global__ __launch_bounds__(256) void kvproj_kernel(const _Float16* __restrict__ s2h,
                                                     const _Float16* __restrict__ WkvT,
                                                     const float* __restrict__ gk,
                                                     _Float16* __restrict__ kf,
                                                     _Float16* __restrict__ vT) {
  const int tile = blockIdx.x;
  const int b    = tile / (N2N / 16);
  const int row0 = (tile - b * (N2N / 16)) * 16;
  const int lane = threadIdx.x & 31;
  const int head = threadIdx.x >> 5;
  const int hgrp = lane >> 4;
  const int lid  = lane & 15;

  const _Float16* arow = s2h + ((size_t)b * N2N + row0 + lid) * CS;
  v8f ck0 = {}, ck1 = {}, cv0 = {}, cv1 = {};
  for (int kk = 0; kk < CS; kk += 32) {
    const v16h a = pack16(arow + kk + hgrp * 16, arow + kk + hgrp * 16 + 8);
    const size_t nb = (size_t)(head * 64 + lid) * CS + kk + hgrp * 16;
    { const _Float16* bp = WkvT + nb;                      const v16h bf = pack16(bp, bp + 8);
      ck0 = __builtin_amdgcn_wmma_f32_16x16x32_f16(false, a, false, bf, (short)0, ck0, false, false); }
    { const _Float16* bp = WkvT + nb + (size_t)16 * CS;    const v16h bf = pack16(bp, bp + 8);
      ck1 = __builtin_amdgcn_wmma_f32_16x16x32_f16(false, a, false, bf, (short)0, ck1, false, false); }
    { const _Float16* bp = WkvT + nb + (size_t)32 * CS;    const v16h bf = pack16(bp, bp + 8);
      cv0 = __builtin_amdgcn_wmma_f32_16x16x32_f16(false, a, false, bf, (short)0, cv0, false, false); }
    { const _Float16* bp = WkvT + nb + (size_t)48 * CS;    const v16h bf = pack16(bp, bp + 8);
      cv1 = __builtin_amdgcn_wmma_f32_16x16x32_f16(false, a, false, bf, (short)0, cv1, false, false); }
  }
  const float gkl = gk[lid], gkh = gk[16 + lid];
#pragma unroll
  for (int r = 0; r < 8; ++r) {
    float p = ck0[r] * ck0[r] + ck1[r] * ck1[r];
    p += __shfl_xor(p, 1); p += __shfl_xor(p, 2); p += __shfl_xor(p, 4); p += __shfl_xor(p, 8);
    const float rinv = rsqrtf(p * (1.0f / 32.0f) + EPSF);
    const int m = hgrp * 8 + r;
    const int row = row0 + m;
    const size_t kbase = (((size_t)b * HN + head) * N2N + row) * (size_t)DN;
    kf[kbase + lid]      = (_Float16)(ck0[r] * rinv * gkl);
    kf[kbase + 16 + lid] = (_Float16)(ck1[r] * rinv * gkh);
    const size_t vbase = ((size_t)b * HN + head) * (size_t)DN * N2N;
    vT[vbase + (size_t)(lid)      * N2N + row] = (_Float16)cv0[r];
    vT[vbase + (size_t)(16 + lid) * N2N + row] = (_Float16)cv1[r];
  }
}

// ---------------------------------------------------------------------------
// Fused flash attention.  grid: B*H*(N1/64), block 128 (wave = 16-row Q tile)
__global__ __launch_bounds__(128) void attn_kernel(const _Float16* __restrict__ qf,
                                                   const _Float16* __restrict__ kf,
                                                   const _Float16* __restrict__ vT,
                                                   const int* __restrict__ mask1,
                                                   const int* __restrict__ mask2,
                                                   _Float16* __restrict__ obuf) {
  __shared__ __align__(16) float    sS[4][16 * 64];
  __shared__ __align__(16) _Float16 sP[4][16 * 64];
  __shared__ float sM[4][16];
  __shared__ float sL[4][16];
  __shared__ float sC[4][16];

  const int nqb  = N1N / 64;
  const int bh   = blockIdx.x / nqb;
  const int qb   = (blockIdx.x - bh * nqb) * 64;
  const int b    = bh / HN;
  const int h    = bh - b * HN;
  const int wave = threadIdx.x >> 5;
  const int lane = threadIdx.x & 31;
  const int hgrp = lane >> 4;
  const int lid  = lane & 15;
  const int row0 = qb + wave * 16;

  // Q A-fragment (rows row0..row0+15, K = D = 32) — loaded once
  const _Float16* qp = qf + ((size_t)bh * N1N + row0 + lid) * DN + hgrp * 16;
  const v16h qa = pack16(qp, qp + 8);

  v8f o0 = {}, o1 = {};
  if (lane < 16) { sM[wave][lane] = -1e30f; sL[wave][lane] = 0.0f; }
  __syncthreads();

  for (int kb = 0; kb < N2N; kb += 64) {
    // S tile: 16x64 scores via 4 WMMAs
#pragma unroll
    for (int t = 0; t < 4; ++t) {
      const _Float16* kp = kf + ((size_t)bh * N2N + kb + t * 16 + lid) * DN + hgrp * 16;
      const v16h kfr = pack16(kp, kp + 8);
      v8f s = {};
      s = __builtin_amdgcn_wmma_f32_16x16x32_f16(false, qa, false, kfr, (short)0, s, false, false);
#pragma unroll
      for (int r = 0; r < 8; ++r) sS[wave][(hgrp * 8 + r) * 64 + t * 16 + lid] = s[r];
    }
    __syncthreads();

    // Online softmax: one row per lane (lanes 0..15 of each wave)
    if (lane < 16) {
      const int m = lane;
      const int m1v = mask1[b * N1N + row0 + m];
      float* Srow = &sS[wave][m * 64];
      const float oldM = sM[wave][m];
      float mx = oldM;
      for (int j = 0; j < 64; ++j) {
        const float mk = (m1v && mask2[b * N2N + kb + j]) ? 0.0f : -INFV;
        const float x = Srow[j] + mk;
        Srow[j] = x;
        mx = fmaxf(mx, x);
      }
      const float corr = __expf(oldM - mx);
      float sum = 0.0f;
      _Float16* Prow = &sP[wave][m * 64];
      for (int j = 0; j < 64; ++j) {
        const float p = __expf(Srow[j] - mx);
        sum += p;
        Prow[j] = (_Float16)p;
      }
      sM[wave][m] = mx;
      sL[wave][m] = sL[wave][m] * corr + sum;
      sC[wave][m] = corr;
    }
    __syncthreads();

    // rescale accumulator by per-row correction
#pragma unroll
    for (int r = 0; r < 8; ++r) {
      const float c = sC[wave][hgrp * 8 + r];
      o0[r] *= c; o1[r] *= c;
    }

    // O += P @ V  (two K=32 steps over the 64-key block, two d-tiles)
#pragma unroll
    for (int ks = 0; ks < 2; ++ks) {
      const _Float16* pp = &sP[wave][lid * 64 + ks * 32 + hgrp * 16];
      const v16h pa = pack16(pp, pp + 8);
      {
        const _Float16* vp = vT + ((size_t)bh * DN + lid) * N2N + kb + ks * 32 + hgrp * 16;
        const v16h vfr = pack16(vp, vp + 8);
        o0 = __builtin_amdgcn_wmma_f32_16x16x32_f16(false, pa, false, vfr, (short)0, o0, false, false);
      }
      {
        const _Float16* vp = vT + ((size_t)bh * DN + 16 + lid) * N2N + kb + ks * 32 + hgrp * 16;
        const v16h vfr = pack16(vp, vp + 8);
        o1 = __builtin_amdgcn_wmma_f32_16x16x32_f16(false, pa, false, vfr, (short)0, o1, false, false);
      }
    }
    __syncthreads();
  }

  // normalize by running sum, write o (B, N1, H*D) f16
#pragma unroll
  for (int r = 0; r < 8; ++r) {
    const int m = hgrp * 8 + r;
    const float li = 1.0f / sL[wave][m];
    const size_t base = ((size_t)b * N1N + row0 + m) * HC + (size_t)h * DN;
    obuf[base + lid]      = (_Float16)(o0[r] * li);
    obuf[base + 16 + lid] = (_Float16)(o1[r] * li);
  }
}

// ---------------------------------------------------------------------------
// out = o @ Wout + b_out (f32 out).  grid: B*N1/16, block 256 (8 waves, 2 n-tiles each)
__global__ __launch_bounds__(256) void outproj_kernel(const _Float16* __restrict__ obuf,
                                                      const _Float16* __restrict__ WoT,
                                                      const float* __restrict__ bout,
                                                      float* __restrict__ out) {
  const int row0 = blockIdx.x * 16;          // rows over flat B*N1
  const int lane = threadIdx.x & 31;
  const int wave = threadIdx.x >> 5;
  const int hgrp = lane >> 4;
  const int lid  = lane & 15;

  const _Float16* arow = obuf + (size_t)(row0 + lid) * HC;
  v8f c0 = {}, c1 = {};
  for (int kk = 0; kk < HC; kk += 32) {
    const v16h a = pack16(arow + kk + hgrp * 16, arow + kk + hgrp * 16 + 8);
    {
      const _Float16* bp = WoT + (size_t)((wave * 2 + 0) * 16 + lid) * HC + kk + hgrp * 16;
      const v16h bf = pack16(bp, bp + 8);
      c0 = __builtin_amdgcn_wmma_f32_16x16x32_f16(false, a, false, bf, (short)0, c0, false, false);
    }
    {
      const _Float16* bp = WoT + (size_t)((wave * 2 + 1) * 16 + lid) * HC + kk + hgrp * 16;
      const v16h bf = pack16(bp, bp + 8);
      c1 = __builtin_amdgcn_wmma_f32_16x16x32_f16(false, a, false, bf, (short)0, c1, false, false);
    }
  }
  const int n0 = (wave * 2 + 0) * 16 + lid;
  const int n1 = (wave * 2 + 1) * 16 + lid;
  const float b0 = bout[n0], b1 = bout[n1];
#pragma unroll
  for (int r = 0; r < 8; ++r) {
    const int m = hgrp * 8 + r;
    out[(size_t)(row0 + m) * HC + n0] = c0[r] + b0;
    out[(size_t)(row0 + m) * HC + n1] = c1[r] + b1;
  }
}

// ---------------------------------------------------------------------------
extern "C" void kernel_launch(void* const* d_in, const int* in_sizes, int n_in,
                              void* d_out, int out_size, void* d_ws, size_t ws_size,
                              hipStream_t stream) {
  (void)in_sizes; (void)n_in; (void)out_size; (void)ws_size;
  const float* s1    = (const float*)d_in[0];
  const float* s2    = (const float*)d_in[1];
  const int*   ridx1 = (const int*)d_in[2];
  const int*   ct1   = (const int*)d_in[3];
  const int*   mask1 = (const int*)d_in[4];
  const int*   mask2 = (const int*)d_in[5];
  const float* Wq    = (const float*)d_in[6];
  const float* Wkv   = (const float*)d_in[7];
  const float* Wout  = (const float*)d_in[8];
  const float* b_out = (const float*)d_in[9];
  const float* gq    = (const float*)d_in[10];
  const float* gk    = (const float*)d_in[11];
  float* out = (float*)d_out;

  char* ws = (char*)d_ws;
  size_t off = 0;
  auto alloc = [&](size_t bytes) -> char* {
    char* p = ws + off;
    off += (bytes + 255) & ~(size_t)255;
    return p;
  };
  _Float16* s1e   = (_Float16*)alloc((size_t)BB * N1N * CS * 2);
  _Float16* s2h   = (_Float16*)alloc((size_t)BB * N2N * CS * 2);
  _Float16* WqT   = (_Float16*)alloc((size_t)CS * HC * 2);
  _Float16* WkvT  = (_Float16*)alloc((size_t)CS * 2 * HC * 2);
  _Float16* WoT   = (_Float16*)alloc((size_t)HC * CS * 2);
  _Float16* qf    = (_Float16*)alloc((size_t)BB * HN * N1N * DN * 2);
  _Float16* kf    = (_Float16*)alloc((size_t)BB * HN * N2N * DN * 2);
  _Float16* vT    = (_Float16*)alloc((size_t)BB * HN * DN * N2N * 2);
  _Float16* obuf  = (_Float16*)alloc((size_t)BB * N1N * HC * 2);

  // 1. input prep (embed + f16 conversions + weight transposes)
  embed_kernel<<<BB * N1N, 256, 0, stream>>>(s1, ridx1, ct1, s1e);
  cvt_kernel<<<(BB * N2N * CS) / 256, 256, 0, stream>>>(s2, s2h, BB * N2N * CS);
  tcvt_kernel<<<(CS * HC) / 256, 256, 0, stream>>>(Wq, WqT, CS, HC);
  tcvt_kernel<<<(CS * 2 * HC) / 256, 256, 0, stream>>>(Wkv, WkvT, CS, 2 * HC);
  tcvt_kernel<<<(HC * CS) / 256, 256, 0, stream>>>(Wout, WoT, HC, CS);

  // 2. projections (+RMS norm, scale folded into Q)
  qproj_kernel<<<(BB * N1N) / 16, 256, 0, stream>>>(s1e, WqT, gq, qf);
  kvproj_kernel<<<(BB * N2N) / 16, 256, 0, stream>>>(s2h, WkvT, gk, kf, vT);

  // 3. fused flash attention
  attn_kernel<<<BB * HN * (N1N / 64), 128, 0, stream>>>(qf, kf, vT, mask1, mask2, obuf);

  // 4. output projection + bias
  outproj_kernel<<<(BB * N1N) / 16, 256, 0, stream>>>(obuf, WoT, b_out, out);
}
